// LinearQKVAttention_26371099198098
// MI455X (gfx1250) — matrix operand order repaired and checked by reading, not compiled, not run on hardware
//
#include <hip/hip_runtime.h>

// ---------------------------------------------------------------------------
// Linear QKV attention for MI455X (gfx1250), wave32 + v_wmma_f32_16x16x32_bf16
//
// res = (softmax_rows(q)/8) @ [ (exp(k)^T @ v) / colsum(exp(k)) ]
// (seq-softmax max-subtraction cancels analytically; inputs are N(0,1) so
//  exp() without max shift is numerically safe in f32)
// ---------------------------------------------------------------------------

#define HEADS 64
#define SEQ   8192
#define HD    64
#define CHUNKS_PER_HEAD 8
#define ROWS_PER_CHUNK (SEQ / CHUNKS_PER_HEAD)   // 1024
#define ITER_ROWS 64
#define PA 72   // LDS pitch (bf16 elems): 144B rows -> 16B aligned, conflict-light

typedef __bf16 bf16_t;
typedef __attribute__((ext_vector_type(8)))  __bf16 v8bf;
typedef __attribute__((ext_vector_type(16))) __bf16 v16bf;
typedef __attribute__((ext_vector_type(8)))  float  v8f;

static __device__ __forceinline__ unsigned short f2bf(float f) {
    unsigned u = __builtin_bit_cast(unsigned, f);
    u += 0x7FFFu + ((u >> 16) & 1u);          // round-to-nearest-even
    return (unsigned short)(u >> 16);
}

// Build a 16-element bf16 WMMA operand from two 16-byte LDS chunks.
// (Both chunk addresses are 16B aligned -> ds_load_b128 each.)
static __device__ __forceinline__ v16bf ld_ab(const unsigned short* lds, int i0, int i1) {
    v8bf lo = *reinterpret_cast<const v8bf*>(lds + i0);
    v8bf hi = *reinterpret_cast<const v8bf*>(lds + i1);
    v16bf r;
#pragma unroll
    for (int i = 0; i < 8; ++i) { r[i] = lo[i]; r[i + 8] = hi[i]; }
    return r;
}

// ---------------------------------------------------------------------------
// Kernel 0: zero the f32 accumulators in workspace
// ---------------------------------------------------------------------------
__global__ void lqkv_ws_zero(float* __restrict__ p, int n) {
    int i = blockIdx.x * blockDim.x + threadIdx.x;
    if (i < n) p[i] = 0.0f;
}

// ---------------------------------------------------------------------------
// Kernel 1: ctx_raw[head][d][e] += sum_n exp(k[n,d]) * v[n,e]
//           colsum[head][d]     += sum_n exp(k[n,d])
// grid = 512 (64 heads x 8 chunks), block = 256 (8 waves)
// Each wave owns 2 of the 16 (m,n) 16x16 tiles of the 64x64 context.
// ---------------------------------------------------------------------------
__global__ void lqkv_ctx_accum(const float* __restrict__ k,
                               const float* __restrict__ v,
                               float* __restrict__ ctx_raw,
                               float* __restrict__ colsum) {
    __shared__ unsigned short At[64 * PA];   // exp(k)^T : [d][n_local]
    __shared__ unsigned short Bt[64 * PA];   // v^T      : [e][n_local]
    __shared__ float csum[64];

    const int tid   = threadIdx.x;
    const int head  = blockIdx.x >> 3;
    const int chunk = blockIdx.x & 7;
    const size_t base = ((size_t)head * SEQ + (size_t)chunk * ROWS_PER_CHUNK) * HD;

    if (tid < 64) csum[tid] = 0.0f;

    const int lane   = tid & 31;
    const int wave   = tid >> 5;
    const int m_tile = wave >> 1;          // 0..3 (d-tile)
    const int nt0    = (wave & 1) * 2;     // 0 or 2 (first e-tile)
    const int h      = lane >> 4;          // half-wave
    const int m      = lane & 15;

    float csum_part = 0.0f;
    v8f acc0 = {}; v8f acc1 = {};

    __syncthreads();

    for (int it = 0; it < ROWS_PER_CHUNK / ITER_ROWS; ++it) {
        const size_t rb = base + (size_t)it * ITER_ROWS * HD;
        // ---- stage 64 rows of k (exp'd) and v, transposed, f32 -> bf16 ----
#pragma unroll
        for (int j = 0; j < (ITER_ROWS * HD) / 256; ++j) {   // 16 iters
            const int idx = tid + 256 * j;
            const int n = idx >> 6;       // row within slab
            const int d = idx & 63;       // channel (== tid&63, fixed)
            const float ek = __expf(k[rb + idx]);
            csum_part += ek;
            At[d * PA + n] = f2bf(ek);
            Bt[d * PA + n] = f2bf(v[rb + idx]);
        }
        __syncthreads();

        // ---- 2 K-steps of WMMA on the 64-row slab ----
#pragma unroll
        for (int ks = 0; ks < 2; ++ks) {
            const int kb = ks * 32;
            // A operand: lane m = row M, K = kb + 16h + [0..15] contiguous
            const int ar = (m_tile * 16 + m) * PA + kb + 16 * h;
            v16bf a = ld_ab(At, ar, ar + 8);
            // B operand: lane m = col N, K = kb + 16h + [0..15] contiguous
            const int br0 = ((nt0 + 0) * 16 + m) * PA + kb + 16 * h;
            const int br1 = ((nt0 + 1) * 16 + m) * PA + kb + 16 * h;
            v16bf b0 = ld_ab(Bt, br0, br0 + 8);
            v16bf b1 = ld_ab(Bt, br1, br1 + 8);
            acc0 = __builtin_amdgcn_wmma_f32_16x16x32_bf16(false, a, false, b0,
                                                           (short)0, acc0, false, false);
            acc1 = __builtin_amdgcn_wmma_f32_16x16x32_bf16(false, a, false, b1,
                                                           (short)0, acc1, false, false);
        }
        __syncthreads();
    }

    // ---- merge partial results ----
    atomicAdd(&csum[tid & 63], csum_part);
    __syncthreads();
    if (tid < 64) atomicAdd(&colsum[head * 64 + tid], csum[tid]);

    float* cr = ctx_raw + (size_t)head * 4096;
#pragma unroll
    for (int r = 0; r < 8; ++r) {   // C/D layout: VGPR r <-> M = r + 8h, lane -> N = m
        const int mrow = m_tile * 16 + r + 8 * h;
        atomicAdd(&cr[mrow * 64 + (nt0 + 0) * 16 + m], acc0[r]);
        atomicAdd(&cr[mrow * 64 + (nt0 + 1) * 16 + m], acc1[r]);
    }
}

// ---------------------------------------------------------------------------
// Kernel 2: ctxT_bf16[head][e][d] = bf16( ctx_raw[head][d][e] * 0.125/colsum[d] )
// (B-operand staging layout for kernel 3; folds the q 1/sqrt(64) scale)
// ---------------------------------------------------------------------------
__global__ void lqkv_ctx_finalize(const float* __restrict__ ctx_raw,
                                  const float* __restrict__ colsum,
                                  unsigned short* __restrict__ ctxT) {
    const int head = blockIdx.x;
    const float* cr = ctx_raw + (size_t)head * 4096;
    const float* cs = colsum + head * 64;
    unsigned short* ct = ctxT + (size_t)head * 4096;
#pragma unroll
    for (int j = 0; j < 16; ++j) {
        const int idx = threadIdx.x + 256 * j;
        const int d = idx >> 6;
        const int e = idx & 63;
        ct[e * 64 + d] = f2bf(cr[idx] * (0.125f / cs[d]));
    }
}

// ---------------------------------------------------------------------------
// Kernel 3: out[n][e] = softmax_row(q[n,:]) @ ctx
// grid = 4096 (64 heads x 64 row-blocks of 128), block = 256 (8 waves)
// Each wave: one 16x64 q tile -> row softmax -> bf16 A tile -> 8 WMMAs.
// ---------------------------------------------------------------------------
__global__ void lqkv_attn_out(const float* __restrict__ q,
                              const unsigned short* __restrict__ ctxT,
                              float* __restrict__ out) {
    __shared__ unsigned short Bt[64 * 64];       // ctx^T [e][d], pitch 64
    __shared__ unsigned short Aq[8][16 * 64];    // per-wave q_sm tiles

    const int tid  = threadIdx.x;
    const int head = blockIdx.x >> 6;
    const int rblk = blockIdx.x & 63;

    // load ctx^T (8 KB) cooperatively
    {
        const uint4* src = reinterpret_cast<const uint4*>(ctxT + (size_t)head * 4096);
        uint4* dst = reinterpret_cast<uint4*>(Bt);
        dst[tid]       = src[tid];
        dst[tid + 256] = src[tid + 256];
    }
    __syncthreads();

    const int wave = tid >> 5;
    const int lane = tid & 31;
    const int row  = lane >> 1;        // 2 lanes per q row
    const int half = lane & 1;         // which 32 channels
    const int r0   = rblk * 128 + wave * 16;

    // ---- load 32 channels of one q row, row softmax via paired lanes ----
    const float4* qp = reinterpret_cast<const float4*>(
        q + ((size_t)head * SEQ + r0 + row) * HD + half * 32);
    float vbuf[32];
#pragma unroll
    for (int j = 0; j < 8; ++j) {
        const float4 f = qp[j];
        vbuf[4 * j] = f.x; vbuf[4 * j + 1] = f.y;
        vbuf[4 * j + 2] = f.z; vbuf[4 * j + 3] = f.w;
    }
    float mx = -3.4e38f;
#pragma unroll
    for (int j = 0; j < 32; ++j) mx = fmaxf(mx, vbuf[j]);
    mx = fmaxf(mx, __shfl_xor(mx, 1, 32));
    float s = 0.0f;
#pragma unroll
    for (int j = 0; j < 32; ++j) { vbuf[j] = __expf(vbuf[j] - mx); s += vbuf[j]; }
    s += __shfl_xor(s, 1, 32);
    const float inv = 1.0f / s;

    // pack bf16 row into A-layout LDS (row-major [m][k], pitch 64)
    unsigned short* aw = &Aq[wave][row * 64 + half * 32];
#pragma unroll
    for (int c = 0; c < 4; ++c) {
        unsigned short p0 = f2bf(vbuf[8 * c + 0] * inv);
        unsigned short p1 = f2bf(vbuf[8 * c + 1] * inv);
        unsigned short p2 = f2bf(vbuf[8 * c + 2] * inv);
        unsigned short p3 = f2bf(vbuf[8 * c + 3] * inv);
        unsigned short p4 = f2bf(vbuf[8 * c + 4] * inv);
        unsigned short p5 = f2bf(vbuf[8 * c + 5] * inv);
        unsigned short p6 = f2bf(vbuf[8 * c + 6] * inv);
        unsigned short p7 = f2bf(vbuf[8 * c + 7] * inv);
        uint4 u;
        u.x = (unsigned)p0 | ((unsigned)p1 << 16);
        u.y = (unsigned)p2 | ((unsigned)p3 << 16);
        u.z = (unsigned)p4 | ((unsigned)p5 << 16);
        u.w = (unsigned)p6 | ((unsigned)p7 << 16);
        *reinterpret_cast<uint4*>(aw + 8 * c) = u;
    }
    // Aq[wave] is wave-private; LDS ops from one wave are in-order -> no barrier.

    const int h = lane >> 4;
    const int m = lane & 15;
    v8f acc[4] = {{}, {}, {}, {}};
#pragma unroll
    for (int ks = 0; ks < 2; ++ks) {
        const int kb = ks * 32;
        const int ar = m * 64 + kb + 16 * h;
        v16bf a = ld_ab(Aq[wave], ar, ar + 8);
#pragma unroll
        for (int nt = 0; nt < 4; ++nt) {
            const int br = (nt * 16 + m) * 64 + kb + 16 * h;
            v16bf b = ld_ab(Bt, br, br + 8);
            acc[nt] = __builtin_amdgcn_wmma_f32_16x16x32_bf16(false, a, false, b,
                                                              (short)0, acc[nt], false, false);
        }
    }

    float* op = out + ((size_t)head * SEQ + r0) * HD;
#pragma unroll
    for (int nt = 0; nt < 4; ++nt) {
#pragma unroll
        for (int r = 0; r < 8; ++r) {
            op[(r + 8 * h) * HD + nt * 16 + m] = acc[nt][r];
        }
    }
}

// ---------------------------------------------------------------------------
extern "C" void kernel_launch(void* const* d_in, const int* in_sizes, int n_in,
                              void* d_out, int out_size, void* d_ws, size_t ws_size,
                              hipStream_t stream) {
    (void)in_sizes; (void)n_in; (void)out_size; (void)ws_size;
    const float* q = (const float*)d_in[0];
    const float* k = (const float*)d_in[1];
    const float* v = (const float*)d_in[2];
    float* out = (float*)d_out;

    // workspace layout
    float* ctx_raw = (float*)d_ws;                       // 64*4096 f32
    float* colsum  = ctx_raw + (size_t)HEADS * 4096;     // 64*64  f32
    unsigned short* ctxT = (unsigned short*)(colsum + HEADS * 64);  // 64*4096 bf16

    const int nzero = HEADS * 4096 + HEADS * 64;
    lqkv_ws_zero<<<(nzero + 255) / 256, 256, 0, stream>>>(ctx_raw, nzero);
    lqkv_ctx_accum<<<HEADS * CHUNKS_PER_HEAD, 256, 0, stream>>>(k, v, ctx_raw, colsum);
    lqkv_ctx_finalize<<<HEADS, 256, 0, stream>>>(ctx_raw, colsum, ctxT);
    lqkv_attn_out<<<HEADS * (SEQ / 128), 256, 0, stream>>>(q, ctxT, out);
}
